// HANRec_59725815218514
// MI455X (gfx1250) — matrix operand order repaired and checked by reading, compiled, not run on hardware
//
#include <hip/hip_runtime.h>
#include <hip/hip_bf16.h>
#include <cstdint>

typedef __attribute__((ext_vector_type(16))) _Float16 v16h;
typedef __attribute__((ext_vector_type(8)))  _Float16 v8h;
typedef __attribute__((ext_vector_type(8)))  float    v8f;

namespace {
constexpr int NUC = 20000, NIC = 20000, EC = 200000, HC = 8, DC = 64, FC = 512;
constexpr int BC = 1024, PC = 8, KC = 8;
}

// ---------- helpers ----------
__device__ __forceinline__ unsigned f2ord(float f) {
    unsigned u = __float_as_uint(f);
    return (u & 0x80000000u) ? ~u : (u | 0x80000000u);
}
__device__ __forceinline__ float ord2f(unsigned u) {
    unsigned b = (u & 0x80000000u) ? (u & 0x7FFFFFFFu) : ~u;
    return __uint_as_float(b);
}

// ---------- fp32 -> fp16 convert (contiguous) ----------
__global__ void han_cvt_f2h(const float* __restrict__ s, _Float16* __restrict__ d, int n) {
    int i = blockIdx.x * blockDim.x + threadIdx.x;
    if (i < n) d[i] = (_Float16)s[i];
}

// ---------- fp32 [K,Ncols] row-major -> fp16 col-major [Ncols][K] ----------
__global__ void han_cvt_transpose(const float* __restrict__ s, _Float16* __restrict__ d,
                                  int Kdim, int Ncols) {
    int i = blockIdx.x * blockDim.x + threadIdx.x;
    if (i >= Kdim * Ncols) return;
    int k = i / Ncols, c = i - k * Ncols;
    d[(size_t)c * Kdim + k] = (_Float16)s[i];
}

// ---------- generic WMMA GEMM: C[M,Ncols] = A[M,K](f16, row stride lda)
//            x Bt (f16 col-major: Bt[c*K + k]); fp32 out, row stride ldc.
// One wave per 16x16 output tile; K multiple of 32; M multiple of 16;
// grid.x * 8 waves == (M/16) * Ntiles exactly.
__global__ void han_gemm_wmma(const _Float16* __restrict__ A, int lda,
                              const _Float16* __restrict__ Bt,
                              float* __restrict__ C, int ldc,
                              int Kdim, int Ntiles) {
    const int wave = threadIdx.x >> 5;
    const int lane = threadIdx.x & 31;
    const int tile = blockIdx.x * 8 + wave;
    const int tm = tile / Ntiles;
    const int tn = tile - tm * Ntiles;
    const int m0 = tm * 16, n0 = tn * 16;
    const int half = lane >> 4, r = lane & 15;

    const _Float16* arow = A + (size_t)(m0 + r) * lda;   // A: lane holds row m0+r
    const _Float16* bcol = Bt + (size_t)(n0 + r) * Kdim; // B: lane holds col n0+r

    v8f acc = {};
    for (int kk = 0; kk < Kdim; kk += 32) {
        // A fragment (16x32): VGPR0-3 = K[kb..kb+7], VGPR4-7 = K[kb+8..kb+15]
        const int kb = kk + half * 16;
        v8h alo = *(const v8h*)(arow + kb);
        v8h ahi = *(const v8h*)(arow + kb + 8);
        v16h a = __builtin_shufflevector(alo, ahi, 0, 1, 2, 3, 4, 5, 6, 7,
                                                   8, 9, 10, 11, 12, 13, 14, 15);
        // B fragment (32x16): lane holds 16 contiguous K halves (col-major)
        v16h b = *(const v16h*)(bcol + kk + half * 16);
        acc = __builtin_amdgcn_wmma_f32_16x16x32_f16(false, a, false, b,
                                                     (short)0, acc, false, false);
    }
    // C/D layout: VGPR j -> M = m0 + j + 8*half, N = n0 + r
#pragma unroll
    for (int j = 0; j < 8; ++j)
        C[(size_t)(m0 + half * 8 + j) * ldc + n0 + r] = acc[j];
}

// ---------- el/er: per (node, head) dot with attention vectors ----------
__global__ void han_elr(const float* __restrict__ f, const float* __restrict__ al,
                        const float* __restrict__ ar, float* __restrict__ el,
                        float* __restrict__ er, int N) {
    int idx = blockIdx.x * blockDim.x + threadIdx.x;
    if (idx >= N * HC) return;
    int n = idx >> 3, h = idx & 7;
    const float* fr = f + (size_t)n * FC + h * DC;
    const float* a1 = al + h * DC;
    const float* a2 = ar + h * DC;
    float s1 = 0.f, s2 = 0.f;
#pragma unroll 8
    for (int d = 0; d < DC; ++d) { float x = fr[d]; s1 += x * a1[d]; s2 += x * a2[d]; }
    el[idx] = s1; er[idx] = s2;
}

// ---------- edge attention + segment max (ordered-uint atomicMax) ----------
__global__ void han_edge_attn_max(const float* __restrict__ el, const float* __restrict__ er,
                                  const int* __restrict__ src, const int* __restrict__ dst,
                                  float* __restrict__ eb, unsigned* __restrict__ mk, int nE) {
    int idx = blockIdx.x * blockDim.x + threadIdx.x;
    if (idx >= nE * HC) return;
    int e = idx >> 3, h = idx & 7;
    int s = src[e], d = dst[e];
    float x = el[s * HC + h] + er[d * HC + h];
    x = x > 0.f ? x : 0.2f * x;                      // leaky_relu(0.2)
    eb[idx] = x;
    atomicMax(&mk[d * HC + h], f2ord(x));
}

// ---------- exp(e - m) + segment sum ----------
__global__ void han_edge_exp_sum(const int* __restrict__ dst, const unsigned* __restrict__ mk,
                                 float* __restrict__ eb, float* __restrict__ sb, int nE) {
    int idx = blockIdx.x * blockDim.x + threadIdx.x;
    if (idx >= nE * HC) return;
    int e = idx >> 3, h = idx & 7;
    int d = dst[e];
    unsigned key = mk[d * HC + h];
    float m = key ? ord2f(key) : 0.f;                // no-edge nodes -> 0 (isfinite fix)
    float ex = expf(eb[idx] - m);
    eb[idx] = ex;
    atomicAdd(&sb[d * HC + h], ex);
}

// ---------- scatter: z[dst] += alpha * f[src]; one wave per edge ----------
__global__ void han_edge_scatter(const int* __restrict__ src, const int* __restrict__ dst,
                                 const float* __restrict__ f, const float* __restrict__ eb,
                                 const float* __restrict__ sb, float* __restrict__ z, int nE) {
    int wave = threadIdx.x >> 5, lane = threadIdx.x & 31;
    int e = blockIdx.x * 8 + wave;
    if (e >= nE) return;
    int s = src[e], d = dst[e];
    const float* fr = f + (size_t)s * FC;
    float* zr = z + (size_t)d * 1024;                // z slab row stride = 2*F
    __builtin_prefetch(fr + lane, 0, 0);             // global_prefetch_b8 into L0/L2
#pragma unroll
    for (int t = 0; t < 16; ++t) {
        int c = lane + 32 * t;
        int h = c >> 6;
        float a = eb[(size_t)e * HC + h] / fmaxf(sb[d * HC + h], 1e-9f);
        atomicAdd(&zr[c], a * fr[c]);
    }
}

// ---------- bias + ELU on z slab ----------
__global__ void han_bias_elu(float* __restrict__ z, const float* __restrict__ bg, int N) {
    int idx = blockIdx.x * blockDim.x + threadIdx.x;
    if (idx >= N * FC) return;
    int n = idx >> 9, c = idx & 511;
    size_t o = (size_t)n * 1024 + c;
    float x = z[o] + bg[c];
    z[o] = x > 0.f ? x : expm1f(x);
}

// ---------- semantic reduction: wsem[m] += mean_n( tanh(t+b1) . v ) ----------
__global__ void han_sem_reduce(const float* __restrict__ t, const float* __restrict__ b1,
                               const float* __restrict__ v, float* __restrict__ wsem,
                               int m, int N) {
    __shared__ float red[128];
    const int tid = threadIdx.x;
    const int n = blockIdx.x;
    red[tid] = tanhf(t[(size_t)n * 128 + tid] + b1[tid]) * v[tid];
    __syncthreads();
    for (int s = 64; s > 0; s >>= 1) { if (tid < s) red[tid] += red[tid + s]; __syncthreads(); }
    if (tid == 0) atomicAdd(&wsem[m], red[0] * (1.0f / (float)N));
}

// ---------- 2-way softmax ----------
__global__ void han_beta(float* __restrict__ wsem) {
    float w0 = wsem[0], w1 = wsem[1];
    float mx = fmaxf(w0, w1);
    float e0 = expf(w0 - mx), e1 = expf(w1 - mx);
    float s = e0 + e1;
    wsem[2] = e0 / s; wsem[3] = e1 / s;
}

// ---------- beta-combine in place: z[:,0,:] = b0*z[:,0,:] + b1*z[:,1,:] ----------
__global__ void han_combine(float* __restrict__ z, const float* __restrict__ wsem, int N) {
    int idx = blockIdx.x * blockDim.x + threadIdx.x;
    if (idx >= N * FC) return;
    int n = idx >> 9, c = idx & 511;
    size_t o = (size_t)n * 1024 + c;
    z[o] = wsem[2] * z[o] + wsem[3] * z[o + FC];
}

// ---------- scoring + broadcast outputs ----------
__global__ void han_score(const float* __restrict__ hu, const float* __restrict__ hi,
                          const int* __restrict__ uid, const int* __restrict__ pos,
                          const int* __restrict__ neg, float* __restrict__ out) {
    __shared__ float ue_s[512];
    __shared__ float red[128];
    const int tid = threadIdx.x;
    const int bp = blockIdx.x;                 // 0..B*P-1
    const int b = bp >> 3;
    const size_t OFF_POS = 0;
    const size_t OFF_NEG = (size_t)BC * PC;                      // 8192
    const size_t OFF_UEN = OFF_NEG + (size_t)BC * PC * KC;       // 73728
    const size_t OFF_PE  = OFF_UEN + (size_t)BC * PC * KC * FC;  // +33.5M
    const size_t OFF_NE  = OFF_PE + (size_t)BC * PC * FC;

    const float* ur = hu + (size_t)uid[b] * 1024;
    for (int i = tid; i < 512; i += 128) ue_s[i] = ur[i];
    __syncthreads();

    const float* pr = hi + (size_t)pos[bp] * 1024;
    float* pe_out = out + OFF_PE + (size_t)bp * FC;
    float part = 0.f;
    for (int i = tid; i < 512; i += 128) { float x = pr[i]; part += ue_s[i] * x; pe_out[i] = x; }
    red[tid] = part; __syncthreads();
    for (int s = 64; s > 0; s >>= 1) { if (tid < s) red[tid] += red[tid + s]; __syncthreads(); }
    if (tid == 0) out[OFF_POS + bp] = red[0];
    __syncthreads();

    for (int k = 0; k < KC; ++k) {
        const float* nr = hi + (size_t)neg[(size_t)bp * KC + k] * 1024;
        size_t base = ((size_t)bp * KC + k) * FC;
        float* ne_out  = out + OFF_NE  + base;
        float* uen_out = out + OFF_UEN + base;
        part = 0.f;
        for (int i = tid; i < 512; i += 128) {
            float x = nr[i]; part += ue_s[i] * x; ne_out[i] = x; uen_out[i] = ue_s[i];
        }
        red[tid] = part; __syncthreads();
        for (int s = 64; s > 0; s >>= 1) { if (tid < s) red[tid] += red[tid + s]; __syncthreads(); }
        if (tid == 0) out[OFF_NEG + (size_t)bp * KC + k] = red[0];
        __syncthreads();
    }
}

extern "C" void kernel_launch(void* const* d_in, const int* in_sizes, int n_in,
                              void* d_out, int out_size, void* d_ws, size_t ws_size,
                              hipStream_t stream) {
    (void)in_sizes; (void)n_in; (void)out_size; (void)ws_size;
    const float* user_embs = (const float*)d_in[0];
    const float* item_embs = (const float*)d_in[1];
    const float* Wg_u = (const float*)d_in[2];
    const float* al_u = (const float*)d_in[3];
    const float* ar_u = (const float*)d_in[4];
    const float* bg_u = (const float*)d_in[5];
    const float* Ws_u = (const float*)d_in[6];
    const float* bs_u = (const float*)d_in[7];
    const float* vs_u = (const float*)d_in[8];
    const float* Wg_i = (const float*)d_in[9];
    const float* al_i = (const float*)d_in[10];
    const float* ar_i = (const float*)d_in[11];
    const float* bg_i = (const float*)d_in[12];
    const float* Ws_i = (const float*)d_in[13];
    const float* bs_i = (const float*)d_in[14];
    const float* vs_i = (const float*)d_in[15];
    const int* src_u = (const int*)d_in[16];
    const int* dst_u = (const int*)d_in[17];
    const int* src_i = (const int*)d_in[18];
    const int* dst_i = (const int*)d_in[19];
    const int* uid = (const int*)d_in[20];
    const int* pos = (const int*)d_in[21];
    const int* neg = (const int*)d_in[22];
    float* out = (float*)d_out;

    // workspace carve-out (256B aligned)
    char* base = (char*)d_ws;
    size_t off = 0;
    auto take = [&](size_t bytes) -> char* {
        char* p = base + off;
        off = (off + bytes + 255) & ~(size_t)255;
        return p;
    };
    float*     zu   = (float*)take((size_t)NUC * 1024 * 4);    // [NU,2,F]
    float*     zi   = (float*)take((size_t)NIC * 1024 * 4);    // [NI,2,F]
    float*     fbuf = (float*)take((size_t)NUC * FC * 4);      // reused per metapath
    _Float16*  h16u = (_Float16*)take((size_t)NUC * DC * 2);
    _Float16*  h16i = (_Float16*)take((size_t)NIC * DC * 2);
    _Float16*  wt   = (_Float16*)take((size_t)FC * DC * 2);    // W^T  [512][64]
    _Float16*  w1t  = (_Float16*)take((size_t)128 * FC * 2);   // W1^T [128][512]
    _Float16*  z16  = (_Float16*)take((size_t)NUC * 1024 * 2); // fp16 z, reused per type
    float*     tbuf = (float*)take((size_t)NUC * 128 * 4);     // z@W1, reused per m
    float*     el   = (float*)take((size_t)NUC * HC * 4);
    float*     er   = (float*)take((size_t)NUC * HC * 4);
    unsigned*  mk   = (unsigned*)take((size_t)NUC * HC * 4);
    float*     sb   = (float*)take((size_t)NUC * HC * 4);
    float*     eb   = (float*)take((size_t)EC * HC * 4);
    float*     wsem = (float*)take(256);

    hipMemsetAsync(zu, 0, (size_t)NUC * 1024 * 4, stream);
    hipMemsetAsync(zi, 0, (size_t)NIC * 1024 * 4, stream);

    han_cvt_f2h<<<(NUC * DC + 255) / 256, 256, 0, stream>>>(user_embs, h16u, NUC * DC);
    han_cvt_f2h<<<(NIC * DC + 255) / 256, 256, 0, stream>>>(item_embs, h16i, NIC * DC);

    for (int t = 0; t < 2; ++t) {
        const int N = (t == 0) ? NUC : NIC;
        const _Float16* h16 = (t == 0) ? h16u : h16i;
        const float* Wg = (t == 0) ? Wg_u : Wg_i;
        const float* al = (t == 0) ? al_u : al_i;
        const float* ar = (t == 0) ? ar_u : ar_i;
        const float* bg = (t == 0) ? bg_u : bg_i;
        const float* Ws = (t == 0) ? Ws_u : Ws_i;
        const float* bs = (t == 0) ? bs_u : bs_i;
        const float* vs = (t == 0) ? vs_u : vs_i;
        const int* src = (t == 0) ? src_u : src_i;
        const int* dst = (t == 0) ? dst_u : dst_i;
        float* z = (t == 0) ? zu : zi;

        for (int mp = 0; mp < 2; ++mp) {
            // W [64,512] -> col-major fp16 [512][64]
            han_cvt_transpose<<<(DC * FC + 255) / 256, 256, 0, stream>>>(
                Wg + (size_t)mp * DC * FC, wt, DC, FC);
            // f = h @ W  (tiles = 1250*32 = 40000; /8 waves = 5000 blocks)
            han_gemm_wmma<<<(N / 16) * 32 / 8, 256, 0, stream>>>(
                h16, DC, wt, fbuf, FC, DC, 32);
            han_elr<<<(N * HC + 255) / 256, 256, 0, stream>>>(
                fbuf, al + (size_t)mp * HC * DC, ar + (size_t)mp * HC * DC, el, er, N);
            hipMemsetAsync(mk, 0, (size_t)N * HC * 4, stream);
            hipMemsetAsync(sb, 0, (size_t)N * HC * 4, stream);
            han_edge_attn_max<<<(EC * HC + 255) / 256, 256, 0, stream>>>(
                el, er, src + (size_t)mp * EC, dst + (size_t)mp * EC, eb, mk, EC);
            han_edge_exp_sum<<<(EC * HC + 255) / 256, 256, 0, stream>>>(
                dst + (size_t)mp * EC, mk, eb, sb, EC);
            han_edge_scatter<<<(EC + 7) / 8, 256, 0, stream>>>(
                src + (size_t)mp * EC, dst + (size_t)mp * EC, fbuf, eb, sb,
                z + (size_t)mp * FC, EC);
            han_bias_elu<<<(N * FC + 255) / 256, 256, 0, stream>>>(
                z + (size_t)mp * FC, bg + (size_t)mp * FC, N);
        }
        // ----- semantic attention (WMMA path) -----
        han_cvt_f2h<<<(N * 1024 + 255) / 256, 256, 0, stream>>>(z, z16, N * 1024);
        han_cvt_transpose<<<(FC * 128 + 255) / 256, 256, 0, stream>>>(Ws, w1t, FC, 128);
        hipMemsetAsync(wsem, 0, 4 * sizeof(float), stream);
        for (int mp = 0; mp < 2; ++mp) {
            // t = z[:,mp,:] @ W1   (tiles = 1250*8 = 10000; /8 = 1250 blocks)
            han_gemm_wmma<<<(N / 16) * 8 / 8, 256, 0, stream>>>(
                z16 + (size_t)mp * FC, 1024, w1t, tbuf, 128, FC, 8);
            han_sem_reduce<<<N, 128, 0, stream>>>(tbuf, bs, vs, wsem, mp, N);
        }
        han_beta<<<1, 1, 0, stream>>>(wsem);
        han_combine<<<(N * FC + 255) / 256, 256, 0, stream>>>(z, wsem, N);
    }

    han_score<<<BC * PC, 128, 0, stream>>>(zu, zi, uid, pos, neg, out);
}
